// attention_41120016891959
// MI455X (gfx1250) — matrix operand order repaired and checked by reading, not compiled, not run on hardware
//
#include <hip/hip_runtime.h>
#include <hip/hip_bf16.h>

#define B_  8
#define T_  1024
#define D_  768
#define H_  12
#define HD_ 64

typedef __attribute__((ext_vector_type(16))) _Float16 v16h;
typedef __attribute__((ext_vector_type(8)))  _Float16 h8;
typedef __attribute__((ext_vector_type(4)))  _Float16 h4;
typedef __attribute__((ext_vector_type(8)))  float    v8f;
typedef __attribute__((ext_vector_type(4)))  float    f4v;
typedef __attribute__((ext_vector_type(4)))  int      v4i;

__device__ inline v8f zero8() {
  v8f z = {0.f, 0.f, 0.f, 0.f, 0.f, 0.f, 0.f, 0.f};
  return z;
}

__device__ inline v8f wmma_f16(v16h a, v16h b, v8f c) {
  // D = A(16x32 f16) * B(32x16 f16) + C(16x16 f32)
  return __builtin_amdgcn_wmma_f32_16x16x32_f16(
      /*neg_a=*/false, a, /*neg_b=*/false, b,
      /*c_mod=*/(short)0, c, /*reuse_a=*/false, /*reuse_b=*/false);
}

// ---- CDNA5 async global->LDS copy (16B per lane), ASYNCcnt-tracked --------
__device__ inline void async_copy_b128(const _Float16* gsrc, _Float16* ldst) {
#if __has_builtin(__builtin_amdgcn_global_load_async_to_lds_b128)
  __builtin_amdgcn_global_load_async_to_lds_b128(
      (__attribute__((address_space(1))) v4i*)(const_cast<_Float16*>(gsrc)),
      (__attribute__((address_space(3))) v4i*)ldst,
      /*offset=*/0, /*cpol=*/0);
#else
  unsigned loff = (unsigned)(uintptr_t)ldst;  // low 32 bits = LDS offset
  asm volatile("global_load_async_to_lds_b128 %0, %1, off"
               :: "v"(loff), "v"(gsrc)
               : "memory");
#endif
}

__device__ inline void wait_async0() {
#if __has_builtin(__builtin_amdgcn_s_wait_asynccnt)
  __builtin_amdgcn_s_wait_asynccnt(0);
#else
  asm volatile("s_wait_asynccnt 0x0" ::: "memory");
#endif
}

// A fragment (16x32, row-major source, leading dim `ld` in elements).
// Lane l (l<16): row=l, K = 0..15 ; lane l>=16: row=l-16, K = 16..31.
__device__ inline v16h load_a_frag(const _Float16* base, int ld) {
  int lane = threadIdx.x & 31;
  int row  = lane & 15;
  int kb   = (lane >> 4) << 4;
  const _Float16* p = base + row * ld + kb;
  h8 lo = *reinterpret_cast<const h8*>(p);
  h8 hi = *reinterpret_cast<const h8*>(p + 8);
  v16h r;
#pragma unroll
  for (int i = 0; i < 8; ++i) { r[i] = lo[i]; r[i + 8] = hi[i]; }
  return r;
}

// B fragment (32x16) from a K-transposed [N][K] buffer: lane l holds column n=l&15,
// K = (l>>4)*16 .. +15 contiguous -> one 32B LDS read.
__device__ inline v16h load_b_frag_t(const _Float16* baseT, int ld) {
  int lane = threadIdx.x & 31;
  int n    = lane & 15;
  int kb   = (lane >> 4) << 4;
  return *reinterpret_cast<const v16h*>(baseT + n * ld + kb);
}

// -------------------------------------------------------------------------
// Fused GEMM: Y = (X @ W + bias) * scale
//   out_h != nullptr : f16 output in head-major layout [B][H][T][HD]  (gridDim.y == H)
//   out_f != nullptr : f32 output row-major [M][D]
// Block tile 128x64, 8 waves (4x2), each wave 32x32 via 4 WMMA accumulators.
// -------------------------------------------------------------------------
__global__ __launch_bounds__(256) void gemm_fused(
    const float* __restrict__ X, const float* __restrict__ W,
    const float* __restrict__ bias, float scale,
    _Float16* __restrict__ out_h, float* __restrict__ out_f)
{
  alignas(128) __shared__ _Float16 As[128][32];  // 8 KB
  alignas(128) __shared__ _Float16 Bs[64][32];   // 4 KB, W tile transposed [n][k]

  const int tid  = threadIdx.x;
  const int wave = tid >> 5;
  const int lane = tid & 31;
  const int wm   = (wave >> 1) * 32;
  const int wn   = (wave & 1) * 32;
  const int m0   = blockIdx.x * 128;
  const int n0   = blockIdx.y * 64;

  v8f acc[2][2];
#pragma unroll
  for (int i = 0; i < 2; ++i)
#pragma unroll
    for (int j = 0; j < 2; ++j) acc[i][j] = zero8();

  for (int k0 = 0; k0 < D_; k0 += 32) {
    // A tile 128x32 (f32 -> f16), float4 loads
#pragma unroll
    for (int it = 0; it < 4; ++it) {
      int q = (tid + it * 256) * 4;
      int r = q >> 5, c = q & 31;
      f4v xv = *reinterpret_cast<const f4v*>(X + (size_t)(m0 + r) * D_ + k0 + c);
      h4 hv;
      hv[0] = (_Float16)xv[0]; hv[1] = (_Float16)xv[1];
      hv[2] = (_Float16)xv[2]; hv[3] = (_Float16)xv[3];
      *reinterpret_cast<h4*>(&As[r][c]) = hv;
    }
    // B tile 32x64 -> transposed Bs[n][k]
#pragma unroll
    for (int it = 0; it < 2; ++it) {
      int q = (tid + it * 256) * 4;
      int k = q >> 6, n = q & 63;
      f4v wv = *reinterpret_cast<const f4v*>(W + (size_t)(k0 + k) * D_ + n0 + n);
#pragma unroll
      for (int j = 0; j < 4; ++j) Bs[n + j][k] = (_Float16)wv[j];
    }
    __syncthreads();

    v16h a0 = load_a_frag(&As[wm][0], 32);
    v16h a1 = load_a_frag(&As[wm + 16][0], 32);
    v16h b0 = load_b_frag_t(&Bs[wn][0], 32);
    v16h b1 = load_b_frag_t(&Bs[wn + 16][0], 32);
    acc[0][0] = wmma_f16(a0, b0, acc[0][0]);
    acc[0][1] = wmma_f16(a0, b1, acc[0][1]);
    acc[1][0] = wmma_f16(a1, b0, acc[1][0]);
    acc[1][1] = wmma_f16(a1, b1, acc[1][1]);
    __syncthreads();
  }

  const int hi = lane >> 4;
  const int nl = lane & 15;
  if (out_f) {
#pragma unroll
    for (int mi = 0; mi < 2; ++mi)
#pragma unroll
      for (int ni = 0; ni < 2; ++ni) {
        int n = n0 + wn + ni * 16 + nl;
        float bv = bias[n];
#pragma unroll
        for (int r = 0; r < 8; ++r) {
          int m = m0 + wm + mi * 16 + hi * 8 + r;
          out_f[(size_t)m * D_ + n] = (acc[mi][ni][r] + bv) * scale;
        }
      }
  } else {
    const int h = blockIdx.y;  // n0 == h * HD_ by construction
#pragma unroll
    for (int mi = 0; mi < 2; ++mi)
#pragma unroll
      for (int ni = 0; ni < 2; ++ni) {
        int hd = wn + ni * 16 + nl;
        float bv = bias[n0 + hd];
#pragma unroll
        for (int r = 0; r < 8; ++r) {
          int m = m0 + wm + mi * 16 + hi * 8 + r;
          int b = m >> 10, t = m & (T_ - 1);
          out_h[(((size_t)(b * H_ + h)) * T_ + t) * HD_ + hd] =
              (_Float16)((acc[mi][ni][r] + bv) * scale);
        }
      }
  }
}

// -------------------------------------------------------------------------
// Gate kernel: gate_a_1[b,h,t] = sigmoid(sum g[0:4]) * (sigmoid(sum g[4:8]) *
// grep[h] - 1) + 2, where g = x_h @ Wg + bg.
// -------------------------------------------------------------------------
__global__ void gates_kernel(const float* __restrict__ X, const float* __restrict__ Wg,
                             const float* __restrict__ bg, const float* __restrict__ grep,
                             float* __restrict__ gate_out)
{
  int idx = blockIdx.x * blockDim.x + threadIdx.x;
  if (idx >= B_ * H_ * T_) return;
  int t = idx & (T_ - 1);
  int h = (idx / T_) % H_;
  int b = idx / (H_ * T_);
  const float* xp = X + ((size_t)(b * T_ + t)) * D_ + h * HD_;
  float acc[8];
#pragma unroll
  for (int j = 0; j < 8; ++j) acc[j] = bg[j];
  for (int i = 0; i < HD_; ++i) {
    float xv = xp[i];
#pragma unroll
    for (int j = 0; j < 8; ++j) acc[j] += xv * Wg[i * 8 + j];
  }
  float sa = acc[0] + acc[1] + acc[2] + acc[3];
  float sb = acc[4] + acc[5] + acc[6] + acc[7];
  float ga = 1.f / (1.f + __expf(-sa));
  float gb = 1.f / (1.f + __expf(-sb));
  gate_out[idx] = ga * (gb * grep[h] - 1.0f) + 2.0f;
}

// -------------------------------------------------------------------------
// Flash attention with fused gated rel_bias.
// Block: one (b,h) x 64 query rows; 4 waves, each owns a 16-row q tile.
// Loops over 64-key chunks; rel_bias streamed exactly once (non-temporal),
// K chunk staged LDS-direct via GLOBAL_LOAD_ASYNC_TO_LDS_B128.
// -------------------------------------------------------------------------
__global__ __launch_bounds__(128) void attn_kernel(
    const _Float16* __restrict__ Q, const _Float16* __restrict__ K,
    const _Float16* __restrict__ V, const float* __restrict__ gate,
    const float* __restrict__ rel_bias, const float* __restrict__ mask,
    float* __restrict__ ctx)
{
  alignas(128) __shared__ _Float16 Ks[64][HD_];     // [key][hd]  8 KB
  alignas(128) __shared__ _Float16 Vt[HD_][64];     // [hd][key]  8 KB
  alignas(128) __shared__ _Float16 Ps[4][16][64];   // per-wave P stage, 8 KB

  const int bh   = blockIdx.x;       // b*H + h
  const int b    = bh / H_;
  const int h    = bh % H_;
  const int wave = threadIdx.x >> 5;
  const int lane = threadIdx.x & 31;
  const int hi   = lane >> 4;
  const int nl   = lane & 15;
  const int q0   = blockIdx.y * 64 + wave * 16;

  const _Float16* Qb = Q + (((size_t)bh) * T_ + q0) * HD_;
  const _Float16* Kb = K + ((size_t)bh) * T_ * HD_;
  const _Float16* Vb = V + ((size_t)bh) * T_ * HD_;
  const float* rb = rel_bias + ((size_t)bh) * T_ * T_;  // [key][q]
  const float* gb = gate + ((size_t)bh) * T_;
  const float* mk = mask + ((size_t)b) * T_;

  // Q fragments for the whole kernel (HD=64 -> two 16x32 A fragments)
  v16h qa0 = load_a_frag(Qb, HD_);
  v16h qa1 = load_a_frag(Qb + 32, HD_);

  float gv[8];
#pragma unroll
  for (int r = 0; r < 8; ++r) gv[r] = gb[q0 + hi * 8 + r];

  float mrow[8], lrow[8];
#pragma unroll
  for (int r = 0; r < 8; ++r) { mrow[r] = -1e30f; lrow[r] = 0.f; }
  v8f o[4];
#pragma unroll
  for (int i = 0; i < 4; ++i) o[i] = zero8();

  for (int kc = 0; kc < T_; kc += 64) {
    // K chunk: async global->LDS (no VGPR round trip, ASYNCcnt-tracked).
    // V chunk: via VGPRs, transposed into Vt on the way in.
#pragma unroll
    for (int it = 0; it < 4; ++it) {
      int e   = (threadIdx.x + it * 128) * 8;   // 0..4088 step 8
      int key = e >> 6, hd = e & 63;
      async_copy_b128(Kb + (size_t)(kc + key) * HD_ + hd, &Ks[key][hd]);
      h8 vv = *reinterpret_cast<const h8*>(Vb + (size_t)(kc + key) * HD_ + hd);
#pragma unroll
      for (int j = 0; j < 8; ++j) Vt[hd + j][key] = vv[j];
    }
    wait_async0();
    __syncthreads();

    // S = Q K^T   (16 q-rows x 64 keys)
    v8f s[4];
#pragma unroll
    for (int nt = 0; nt < 4; ++nt) {
      v8f z = zero8();
      z     = wmma_f16(qa0, load_b_frag_t(&Ks[nt * 16][0],  HD_), z);
      s[nt] = wmma_f16(qa1, load_b_frag_t(&Ks[nt * 16][32], HD_), z);
    }

    // fused gate * rel_bias^T + mask, track chunk row-max.
    // rel_bias is streamed once -> non-temporal loads keep L2 for K/V reuse.
    float lm[8];
#pragma unroll
    for (int r = 0; r < 8; ++r) lm[r] = -1e30f;
#pragma unroll
    for (int nt = 0; nt < 4; ++nt) {
      int key = kc + nt * 16 + nl;
      float mval = mk[key];
      const float* rp = rb + (size_t)key * T_ + q0 + hi * 8;
      if (kc + 64 < T_) __builtin_prefetch(rp + (size_t)64 * T_, 0, 1);
      f4v r0 = __builtin_nontemporal_load(reinterpret_cast<const f4v*>(rp));
      f4v r1 = __builtin_nontemporal_load(reinterpret_cast<const f4v*>(rp + 4));
#pragma unroll
      for (int r = 0; r < 8; ++r) {
        float rbv = (r < 4) ? r0[r] : r1[r - 4];
        float sv = s[nt][r] + gv[r] * rbv + mval;
        s[nt][r] = sv;
        lm[r] = fmaxf(lm[r], sv);
      }
    }
#pragma unroll
    for (int r = 0; r < 8; ++r)
#pragma unroll
      for (int off = 1; off < 16; off <<= 1)
        lm[r] = fmaxf(lm[r], __shfl_xor(lm[r], off, 32));

    // online softmax update
    float ls[8];
#pragma unroll
    for (int r = 0; r < 8; ++r) {
      float mnew  = fmaxf(mrow[r], lm[r]);
      float alpha = __expf(mrow[r] - mnew);
      mrow[r] = mnew;
      float sum = 0.f;
#pragma unroll
      for (int nt = 0; nt < 4; ++nt) {
        float p = __expf(s[nt][r] - mnew);
        s[nt][r] = p;
        sum += p;
      }
      ls[r] = sum;
      lrow[r] *= alpha;
#pragma unroll
      for (int nt = 0; nt < 4; ++nt) o[nt][r] *= alpha;
    }
#pragma unroll
    for (int r = 0; r < 8; ++r) {
#pragma unroll
      for (int off = 1; off < 16; off <<= 1)
        ls[r] += __shfl_xor(ls[r], off, 32);
      lrow[r] += ls[r];
    }

    // stage P (f16) into per-wave LDS, feed back as WMMA A fragments
#pragma unroll
    for (int nt = 0; nt < 4; ++nt)
#pragma unroll
      for (int r = 0; r < 8; ++r)
        Ps[wave][hi * 8 + r][nt * 16 + nl] = (_Float16)s[nt][r];

#pragma unroll
    for (int kk = 0; kk < 2; ++kk) {
      v16h pa = load_a_frag(&Ps[wave][0][kk * 32], 64);
#pragma unroll
      for (int nt = 0; nt < 4; ++nt)
        o[nt] = wmma_f16(pa, load_b_frag_t(&Vt[nt * 16][kk * 32], 64), o[nt]);
    }
    __syncthreads();
  }

  // ctx[b][q][h*HD + hd] = O / l   (row-major for the output projection GEMM)
#pragma unroll
  for (int nt = 0; nt < 4; ++nt) {
    int hd = nt * 16 + nl;
#pragma unroll
    for (int r = 0; r < 8; ++r) {
      int q = q0 + hi * 8 + r;
      ctx[((size_t)(b * T_ + q)) * D_ + h * HD_ + hd] = o[nt][r] / lrow[r];
    }
  }
}

// -------------------------------------------------------------------------
extern "C" void kernel_launch(void* const* d_in, const int* in_sizes, int n_in,
                              void* d_out, int out_size, void* d_ws, size_t ws_size,
                              hipStream_t stream) {
  (void)in_sizes; (void)n_in; (void)out_size; (void)ws_size;
  const float* x         = (const float*)d_in[0];
  const float* attn_mask = (const float*)d_in[1];
  const float* rel_bias  = (const float*)d_in[2];
  const float* Wq = (const float*)d_in[3];
  const float* bq = (const float*)d_in[4];
  const float* Wk = (const float*)d_in[5];
  const float* bk = (const float*)d_in[6];
  const float* Wv = (const float*)d_in[7];
  const float* bv = (const float*)d_in[8];
  const float* Wo = (const float*)d_in[9];
  const float* bo = (const float*)d_in[10];
  const float* Wg = (const float*)d_in[11];
  const float* bg = (const float*)d_in[12];
  const float* grep = (const float*)d_in[13];

  char* ws = (char*)d_ws;
  const size_t qkv_bytes  = (size_t)B_ * H_ * T_ * HD_ * sizeof(_Float16); // 12.6 MB
  const size_t gate_bytes = (size_t)B_ * H_ * T_ * sizeof(float);
  _Float16* Qw = (_Float16*)(ws);
  _Float16* Kw = (_Float16*)(ws + qkv_bytes);
  _Float16* Vw = (_Float16*)(ws + 2 * qkv_bytes);
  float* gatew = (float*)(ws + 3 * qkv_bytes);
  float* ctx   = (float*)(ws + 3 * qkv_bytes + gate_bytes);

  dim3 gblk(64, 12);  // 8192/128 x 768/64
  const float scaling = 0.125f;  // HD^-0.5, folded into Q

  gemm_fused<<<gblk, 256, 0, stream>>>(x, Wq, bq, scaling, Qw, nullptr);
  gemm_fused<<<gblk, 256, 0, stream>>>(x, Wk, bk, 1.0f, Kw, nullptr);
  gemm_fused<<<gblk, 256, 0, stream>>>(x, Wv, bv, 1.0f, Vw, nullptr);
  gates_kernel<<<(B_ * H_ * T_ + 255) / 256, 256, 0, stream>>>(x, Wg, bg, grep, gatew);
  attn_kernel<<<dim3(B_ * H_, T_ / 64), 128, 0, stream>>>(
      Qw, Kw, Vw, gatew, rel_bias, attn_mask, ctx);
  gemm_fused<<<gblk, 256, 0, stream>>>(ctx, Wo, bo, 1.0f, nullptr, (float*)d_out);
}
